// RoPEMultiHeadedAttention_23356032156263
// MI455X (gfx1250) — matrix operand order repaired and checked by reading, not compiled, not run on hardware
//
#include <hip/hip_runtime.h>
#include <hip/hip_bf16.h>

// ---------------------------------------------------------------------------
// RoPE Multi-Headed Attention for MI455X (gfx1250, wave32, WMMA bf16).
// - bf16 V_WMMA_F32_16X16X32_BF16 with fp32 accumulation everywhere.
// - Fused attention: never materializes the S x S score tensors (512 MB each in fp32)
//   to HBM; rotated scores recomputed in pass 2.
// - Pass-1 key tiles staged block-cooperatively via per-lane async-LDS loads
//   (GLOBAL_LOAD_ASYNC_TO_LDS_B128 + s_wait_asynccnt).
// - Pass-2 key tiles staged via the Tensor Data Mover (TENSOR_LOAD_TO_LDS,
//   descriptor per ISA D# layout, s_wait_tensorcnt), issued by one wave.
// ---------------------------------------------------------------------------

typedef __attribute__((ext_vector_type(16))) __bf16 v16bf;
typedef __attribute__((ext_vector_type(8)))  __bf16 v8bf;
typedef __attribute__((ext_vector_type(8)))  float  v8f;
typedef __attribute__((ext_vector_type(4)))  unsigned int v4u;
typedef __attribute__((ext_vector_type(8)))  int v8i;
typedef __attribute__((ext_vector_type(4)))  int v4i;

union BF16x16 { v16bf v; v8bf h[2]; };

static constexpr int kB  = 2;
static constexpr int kS  = 2048;
static constexpr int kD  = 1024;     // model dim == H*HD
static constexpr int kH  = 16;
static constexpr int kHD = 64;
static constexpr int kM  = kB * kS;  // 4096 flattened GEMM rows

__device__ __forceinline__ v8f wmma_bf16(v16bf a, v16bf b, v8f c) {
  return __builtin_amdgcn_wmma_f32_16x16x32_bf16(false, a, false, b,
                                                 (short)0, c, false, false);
}

// Per-lane async copy of 16 bytes global -> LDS (ASYNCcnt-tracked).
__device__ __forceinline__ void async_b128_to_lds(uint32_t lds_byte_off,
                                                  const void* gptr) {
  uint64_t ga = (uint64_t)(uintptr_t)gptr;
  asm volatile("global_load_async_to_lds_b128 %0, %1, off"
               :: "v"(lds_byte_off), "v"(ga)
               : "memory");
}
__device__ __forceinline__ void wait_asynccnt0() {
  asm volatile("s_wait_asynccnt 0x0" ::: "memory");
}

// TDM: 2D tile (rows x 64 bf16 elems, row stride 64 elems) global -> LDS.
// Descriptor per CDNA5 ISA D# groups 0/1 (8.3, 8.4); groups 2/3 zero (2D).
__device__ __forceinline__ void tdm_load_tile_2d(uint32_t lds_byte_off,
                                                 const void* gptr,
                                                 uint32_t rows) {
  uint64_t ga = (uint64_t)(uintptr_t)gptr;
  v4u g0;
  g0[0] = 1u;                                   // count=1 (valid user D#)
  g0[1] = lds_byte_off;                         // lds_addr (bytes)
  g0[2] = (uint32_t)ga;                         // global_addr[31:0]
  g0[3] = (uint32_t)(ga >> 32) | (2u << 30);    // global_addr[56:32], type=2
  v8i g1;
  g1[0] = (int)(1u << 16);                      // wg_mask=0, data_size=1 (2B)
  g1[1] = (int)(64u << 16);                     // tensor_dim0 = 64 (lo16)
  g1[2] = (int)(rows << 16);                    // tensor_dim1 = rows (lo16)
  g1[3] = (int)(64u << 16);                     // tile_dim0 = 64
  g1[4] = (int)rows;                            // tile_dim1 = rows, dim2=0
  g1[5] = 64;                                   // tensor_dim0_stride = 64
  g1[6] = 0;                                    // stride hi / dim1_stride lo
  g1[7] = 0;
  v4i g2 = {0, 0, 0, 0};
  v4i g3 = {0, 0, 0, 0};
#if defined(__clang_major__) && (__clang_major__ >= 23)
  v8i g4 = {0, 0, 0, 0, 0, 0, 0, 0};
  __builtin_amdgcn_tensor_load_to_lds(g0, g1, g2, g3, g4, 0);
#else
  __builtin_amdgcn_tensor_load_to_lds(g0, g1, g2, g3, 0);
#endif
}

// ---------------------------------------------------------------------------
// Weight transpose + fp32 -> bf16 convert: dst[n*K + k] = src[k*N + n]
// ---------------------------------------------------------------------------
__global__ __launch_bounds__(256)
void convert_transpose_kernel(const float* __restrict__ src,
                              __bf16* __restrict__ dst, int K, int N) {
  int idx = blockIdx.x * blockDim.x + threadIdx.x;
  if (idx < K * N) {
    int n = idx / K;
    int k = idx - n * K;
    dst[idx] = (__bf16)src[(size_t)k * N + n];
  }
}

// ---------------------------------------------------------------------------
// GEMM: C[16x16 per wave] = X[4096 x 1024] * W^T-layout[1024 x 1024] + bias.
// MODE 0: Q -> plain + RoPE'd [B,H,S,64]; 1: K -> same; 2: V -> [B,H,64,S];
// MODE 3: out-projection (bf16 A), fp32 result.
// ---------------------------------------------------------------------------
template <int MODE>
__global__ __launch_bounds__(256)
void proj_kernel(const float* __restrict__ Xf, const __bf16* __restrict__ Xb,
                 const __bf16* __restrict__ WT, const float* __restrict__ bias,
                 __bf16* __restrict__ outP, __bf16* __restrict__ outR,
                 float* __restrict__ outF) {
  const int wave = threadIdx.x >> 5;
  const int lane = threadIdx.x & 31;
  const int lh   = lane >> 4;
  const int col  = lane & 15;

  const int rowTile = blockIdx.x;              // 0..255 (16 rows each)
  const int colTile = blockIdx.y * 8 + wave;   // 0..63  (16 cols each)
  const int arowg   = rowTile * 16 + col;      // A row (M = lane%16)
  const int ncol    = colTile * 16 + col;      // C column (N = lane%16)

  v8f c = {0.f, 0.f, 0.f, 0.f, 0.f, 0.f, 0.f, 0.f};

  for (int kc = 0; kc < kD / 32; ++kc) {
    BF16x16 a;
    if (MODE < 3) {
      const float* ap = Xf + (size_t)arowg * kD + kc * 32;
      __builtin_prefetch((const void*)(ap + 32), 0, 0);  // next K chunk
      v8f f0 = *(const v8f*)(ap + lh * 16);
      v8f f1 = *(const v8f*)(ap + lh * 16 + 8);
#pragma unroll
      for (int i = 0; i < 8; ++i) {
        a.h[0][i] = (__bf16)f0[i];
        a.h[1][i] = (__bf16)f1[i];
      }
    } else {
      const __bf16* ap = Xb + (size_t)arowg * kD + kc * 32;
      a.h[0] = *(const v8bf*)(ap + lh * 16);
      a.h[1] = *(const v8bf*)(ap + lh * 16 + 8);
    }
    const __bf16* bp = WT + (size_t)ncol * kD + kc * 32 + lh * 16;
    BF16x16 bb;
    bb.h[0] = *(const v8bf*)(bp);
    bb.h[1] = *(const v8bf*)(bp + 8);
    c = wmma_bf16(a.v, bb.v, c);
  }

  const float bco = bias[ncol];

  if (MODE == 3) {
#pragma unroll
    for (int r = 0; r < 8; ++r) {
      int mg = rowTile * 16 + r + 8 * lh;
      outF[(size_t)mg * kD + ncol] = c[r] + bco;
    }
    return;
  }

  const int hh = ncol >> 6;
  const int d  = ncol & 63;

  if (MODE == 2) {  // V transposed: [bh][d][s]
#pragma unroll
    for (int r = 0; r < 8; ++r) {
      int mg = rowTile * 16 + r + 8 * lh;
      int b  = mg >> 11, s = mg & (kS - 1);
      outP[(((size_t)b * kH + hh) * kHD + d) * kS + s] = (__bf16)(c[r] + bco);
    }
    return;
  }

  // Q / K: plain + interleaved-pair RoPE (base 1000)
  const float theta = __powf(1000.0f, -(float)(d & ~1) * (1.0f / 64.0f));
#pragma unroll
  for (int r = 0; r < 8; ++r) {
    int mg = rowTile * 16 + r + 8 * lh;
    int b  = mg >> 11, s = mg & (kS - 1);
    float val = c[r] + bco;
    size_t base = (((size_t)b * kH + hh) * kS + s) * kHD + d;
    outP[base] = (__bf16)val;
    float pv = __shfl_xor(val, 1, 32);   // pair partner (d^1) in lane^1
    float sn, cs;
    __sincosf((float)s * theta, &sn, &cs);
    float rv = (d & 1) ? (val * cs + pv * sn) : (val * cs - pv * sn);
    outR[base] = (__bf16)rv;
  }
}

// ---------------------------------------------------------------------------
// Fused attention. One wave owns (b, h, 16 query rows); all 8 waves of a
// block share (b,h): K tiles staged once per block into LDS (pass 1: async
// loads; pass 2: TDM) and consumed by all waves.
// ---------------------------------------------------------------------------
__global__ __launch_bounds__(256)
void attn_kernel(const __bf16* __restrict__ qn, const __bf16* __restrict__ kn,
                 const __bf16* __restrict__ qr, const __bf16* __restrict__ kr,
                 const __bf16* __restrict__ vT, __bf16* __restrict__ z) {
  // pass1: [0..1023] = kr tile (16x64), [1024..2047] = kn tile (16x64)
  // pass2: [0..2047] = kr tile (32x64)
  __shared__ __align__(32) __bf16 ldsStage[32 * 64];   // 4 KB
  __shared__ __align__(32) __bf16 ldsE[8][16 * 32];    // 8 KB (1 e-tile/wave)

  const int tid  = threadIdx.x;
  const int wave = tid >> 5;
  const int lane = tid & 31;
  const int lh   = lane >> 4;
  const int col  = lane & 15;

  const int w  = blockIdx.x * 8 + wave;   // 0..4095
  const int qt = w & 127;                 // query tile within S
  const int bh = w >> 7;                  // b*16 + h  (same for whole block)
  const size_t hbase = (size_t)bh * kS * kHD;
  const __bf16* qrp = qr + hbase;
  const __bf16* qnp = qn + hbase;
  const __bf16* krp = kr + hbase;
  const __bf16* knp = kn + hbase;
  const __bf16* vtp = vT + hbase;         // [64][S] for this (b,h)

  const uint32_t stageBase = (uint32_t)(uintptr_t)&ldsStage[0];

  // Resident A fragments (rotated + plain queries)
  BF16x16 aqr[2], aqn[2];
  {
    const __bf16* q0 = qrp + (size_t)(qt * 16 + col) * kHD;
    const __bf16* q1 = qnp + (size_t)(qt * 16 + col) * kHD;
#pragma unroll
    for (int cc = 0; cc < 2; ++cc) {
      aqr[cc].h[0] = *(const v8bf*)(q0 + cc * 32 + lh * 16);
      aqr[cc].h[1] = *(const v8bf*)(q0 + cc * 32 + lh * 16 + 8);
      aqn[cc].h[0] = *(const v8bf*)(q1 + cc * 32 + lh * 16);
      aqn[cc].h[1] = *(const v8bf*)(q1 + cc * 32 + lh * 16 + 8);
    }
  }

  const v8f vzero = {0.f, 0.f, 0.f, 0.f, 0.f, 0.f, 0.f, 0.f};
  float den[8], mx[8];
#pragma unroll
  for (int r = 0; r < 8; ++r) { den[r] = 0.f; mx[r] = -3.0e38f; }

  // Per-thread staging role, pass1: threads 0..127 copy kr, 128..255 copy kn.
  const int p1row = (tid & 127) >> 3;      // 0..15 key row within tile
  const int p1seg = tid & 7;               // 8 x 16B segments per 128B row
  const int p1arr = tid >> 7;              // 0 = kr, 1 = kn
  const uint32_t p1lds =
      stageBase + (uint32_t)(p1arr * 2048 + p1row * 128 + p1seg * 16);

  // ---------------- pass 1: den + masked max of rotated scores ------------
  for (int kt = 0; kt < kS / 16; ++kt) {
    const __bf16* src = (p1arr == 0) ? krp : knp;
    async_b128_to_lds(p1lds,
                      src + (size_t)(kt * 16 + p1row) * kHD + p1seg * 8);
    wait_asynccnt0();
    __syncthreads();   // tile ready

    const __bf16* rrow = &ldsStage[col * 64 + lh * 16];          // kr
    const __bf16* nrow = &ldsStage[1024 + col * 64 + lh * 16];   // kn
    BF16x16 br0, br1, bn0, bn1;
    br0.h[0] = *(const v8bf*)(rrow);      br0.h[1] = *(const v8bf*)(rrow + 8);
    br1.h[0] = *(const v8bf*)(rrow + 32); br1.h[1] = *(const v8bf*)(rrow + 40);
    bn0.h[0] = *(const v8bf*)(nrow);      bn0.h[1] = *(const v8bf*)(nrow + 8);
    bn1.h[0] = *(const v8bf*)(nrow + 32); bn1.h[1] = *(const v8bf*)(nrow + 40);

    v8f sr = vzero, sp = vzero;
    sr = wmma_bf16(aqr[0].v, br0.v, sr);
    sr = wmma_bf16(aqr[1].v, br1.v, sr);
    sp = wmma_bf16(aqn[0].v, bn0.v, sp);
    sp = wmma_bf16(aqn[1].v, bn1.v, sp);

    const int key = kt * 16 + col;
#pragma unroll
    for (int r = 0; r < 8; ++r) {
      den[r] += __expf(sp[r] * 0.125f);           // unmasked full sum (ref)
      int q = qt * 16 + r + 8 * lh;
      float s = sr[r] * 0.125f;
      if (key <= q) mx[r] = fmaxf(mx[r], s);      // causal mask
    }
    __syncthreads();   // all waves done reading tile; safe to overwrite
  }
  // reduce across the 16 lanes of each half (xor 1/2/4/8 stays in-half)
#pragma unroll
  for (int off = 8; off; off >>= 1) {
#pragma unroll
    for (int r = 0; r < 8; ++r) {
      den[r] += __shfl_xor(den[r], off, 32);
      mx[r]   = fmaxf(mx[r], __shfl_xor(mx[r], off, 32));
    }
  }
  float rden[8], amax[8];
#pragma unroll
  for (int r = 0; r < 8; ++r) {
    rden[r] = 1.0f / den[r];
    amax[r] = __expf(mx[r]) * rden[r];
  }

  // ---------------- pass 2: alpha and alpha @ V ---------------------------
  v8f acc[4] = {vzero, vzero, vzero, vzero};
  float lsum[8];
#pragma unroll
  for (int r = 0; r < 8; ++r) lsum[r] = 0.f;

  for (int kt2 = 0; kt2 < kS / 32; ++kt2) {
    if (wave == 0) {   // one TDM op stages the whole 32x64 kr tile
      tdm_load_tile_2d(stageBase, krp + (size_t)kt2 * 32 * kHD, 32u);
      __builtin_amdgcn_s_wait_tensorcnt((short)0);
    }
    __syncthreads();   // tile ready for all waves

#pragma unroll
    for (int s2 = 0; s2 < 2; ++s2) {
      const __bf16* rrow = &ldsStage[(s2 * 16 + col) * 64 + lh * 16];
      BF16x16 br0, br1;
      br0.h[0] = *(const v8bf*)(rrow);      br0.h[1] = *(const v8bf*)(rrow + 8);
      br1.h[0] = *(const v8bf*)(rrow + 32); br1.h[1] = *(const v8bf*)(rrow + 40);
      v8f sr = vzero;
      sr = wmma_bf16(aqr[0].v, br0.v, sr);
      sr = wmma_bf16(aqr[1].v, br1.v, sr);
      const int key = kt2 * 32 + s2 * 16 + col;
#pragma unroll
      for (int r = 0; r < 8; ++r) {
        int q = qt * 16 + r + 8 * lh;
        float a = __expf(sr[r] * 0.125f) * rden[r];
        float e = (key <= q) ? __expf(a - amax[r]) : 0.f;
        lsum[r] += e;
        ldsE[wave][(r + 8 * lh) * 32 + s2 * 16 + col] = (__bf16)e;
      }
    }
    __syncthreads();   // kr-tile readers done; e-tiles committed

    BF16x16 ae;        // e-tile as A fragment (K = 32 keys)
    const __bf16* ep = &ldsE[wave][col * 32];
    ae.h[0] = *(const v8bf*)(ep + lh * 16);
    ae.h[1] = *(const v8bf*)(ep + lh * 16 + 8);
#pragma unroll
    for (int dt = 0; dt < 4; ++dt) {
      const __bf16* vrow =
          vtp + (size_t)(dt * 16 + col) * kS + kt2 * 32 + lh * 16;
      BF16x16 bv;
      bv.h[0] = *(const v8bf*)(vrow);
      bv.h[1] = *(const v8bf*)(vrow + 8);
      acc[dt] = wmma_bf16(ae.v, bv.v, acc[dt]);
    }
  }

#pragma unroll
  for (int off = 8; off; off >>= 1)
#pragma unroll
    for (int r = 0; r < 8; ++r) lsum[r] += __shfl_xor(lsum[r], off, 32);

  const int b = bh >> 4, hh = bh & 15;
#pragma unroll
  for (int dt = 0; dt < 4; ++dt) {
    int d = dt * 16 + col;
#pragma unroll
    for (int r = 0; r < 8; ++r) {
      int s = qt * 16 + r + 8 * lh;
      float zv = acc[dt][r] / lsum[r];
      z[((size_t)(b * kS + s)) * kD + hh * kHD + d] = (__bf16)zv;
    }
  }
}

// ---------------------------------------------------------------------------
extern "C" void kernel_launch(void* const* d_in, const int* in_sizes, int n_in,
                              void* d_out, int out_size, void* d_ws,
                              size_t ws_size, hipStream_t stream) {
  (void)in_sizes; (void)n_in; (void)out_size; (void)ws_size;
  const float* query = (const float*)d_in[0];
  const float* keyx  = (const float*)d_in[1];
  const float* value = (const float*)d_in[2];
  // d_in[3] = causal mask (tril) — encoded directly in the kernel
  const float* Wq = (const float*)d_in[4];
  const float* bq = (const float*)d_in[5];
  const float* Wk = (const float*)d_in[6];
  const float* bk = (const float*)d_in[7];
  const float* Wv = (const float*)d_in[8];
  const float* bv = (const float*)d_in[9];
  const float* Wo = (const float*)d_in[10];
  const float* bo = (const float*)d_in[11];

  __bf16* ws = (__bf16*)d_ws;
  size_t off = 0;
  const size_t wsz = (size_t)kD * kD;            // 1M elems per weight
  const size_t qsz = (size_t)kB * kH * kS * kHD; // 4M elems per tensor
  __bf16* WqT = ws + off; off += wsz;
  __bf16* WkT = ws + off; off += wsz;
  __bf16* WvT = ws + off; off += wsz;
  __bf16* WoT = ws + off; off += wsz;
  __bf16* qnb = ws + off; off += qsz;   // plain Q   [B,H,S,64]
  __bf16* knb = ws + off; off += qsz;   // plain K
  __bf16* qrb = ws + off; off += qsz;   // RoPE'd Q
  __bf16* krb = ws + off; off += qsz;   // RoPE'd K
  __bf16* vtb = ws + off; off += qsz;   // V transposed [B,H,64,S]
  __bf16* zbf = ws + off; off += qsz;   // attention output [B,S,1024]

  const int wThreads = 256, wBlocks = (int)(wsz / wThreads);
  convert_transpose_kernel<<<wBlocks, wThreads, 0, stream>>>(Wq, WqT, kD, kD);
  convert_transpose_kernel<<<wBlocks, wThreads, 0, stream>>>(Wk, WkT, kD, kD);
  convert_transpose_kernel<<<wBlocks, wThreads, 0, stream>>>(Wv, WvT, kD, kD);
  convert_transpose_kernel<<<wBlocks, wThreads, 0, stream>>>(Wo, WoT, kD, kD);

  dim3 pgrid(kM / 16, (kD / 16) / 8);   // 256 x 8 blocks, 8 waves each
  proj_kernel<0><<<pgrid, 256, 0, stream>>>(query, nullptr, WqT, bq, qnb, qrb,
                                            nullptr);
  proj_kernel<1><<<pgrid, 256, 0, stream>>>(keyx, nullptr, WkT, bk, knb, krb,
                                            nullptr);
  proj_kernel<2><<<pgrid, 256, 0, stream>>>(value, nullptr, WvT, bv, vtb,
                                            nullptr, nullptr);

  attn_kernel<<<(kB * kH * (kS / 16)) / 8, 256, 0, stream>>>(qnb, knb, qrb,
                                                             krb, vtb, zbf);

  proj_kernel<3><<<pgrid, 256, 0, stream>>>(nullptr, zbf, WoT, bo, nullptr,
                                            nullptr, (float*)d_out);
}
